// LCNECortexLSTM_4303557230935
// MI455X (gfx1250) — matrix operand order repaired and checked by reading, not compiled, not run on hardware
//
#include <hip/hip_runtime.h>
#include <hip/hip_bf16.h>
#include <math.h>

// ---------------------------------------------------------------------------
// LC-NE-Cortex LSTM step on MI455X (gfx1250).
// bf16 WMMA (f32 accumulate), async-LDS A-tile staging, fused BN statistics.
// B=65536, D=H=256.
// ---------------------------------------------------------------------------

typedef __bf16 bf16;
typedef bf16  bf16x16 __attribute__((ext_vector_type(16)));
typedef float f32x8   __attribute__((ext_vector_type(8)));

#define HDIM 256
#define NROWS 65536
static constexpr size_t BH = (size_t)NROWS * HDIM;   // 16,777,216 elements

__device__ __forceinline__ f32x8 zero8() {
  f32x8 z;
#pragma unroll
  for (int i = 0; i < 8; ++i) z[i] = 0.0f;
  return z;
}

__device__ __forceinline__ f32x8 wmma_bf16(bf16x16 a, bf16x16 b, f32x8 c) {
  return __builtin_amdgcn_wmma_f32_16x16x32_bf16(false, a, false, b, (short)0, c,
                                                 false, false);
}

__device__ __forceinline__ float sigmoidf_(float x) {
  return 1.0f / (1.0f + __expf(-x));
}

// ---------------------------------------------------------------------------
// Async staging: copy one contiguous 32x256-f32 tile (32 KB) global -> LDS.
// 256 threads x 128 B each, 8x GLOBAL_LOAD_ASYNC_TO_LDS_B128 (GVS mode:
// saddr = tile base, vaddr = byte offset applied to BOTH global and LDS sides,
// IOFFSET covers the 8 sub-chunks).  Tracked by ASYNCcnt.
// ---------------------------------------------------------------------------
__device__ __forceinline__ void async_tile_load(const float* __restrict__ g,
                                                void* smem) {
  const unsigned voff = (unsigned)threadIdx.x * 128u;
  const unsigned lds = (unsigned)(uintptr_t)smem + voff;  // low 32 bits = LDS offset
  asm volatile(
      "global_load_async_to_lds_b128 %0, %1, %2 offset:0\n\t"
      "global_load_async_to_lds_b128 %0, %1, %2 offset:16\n\t"
      "global_load_async_to_lds_b128 %0, %1, %2 offset:32\n\t"
      "global_load_async_to_lds_b128 %0, %1, %2 offset:48\n\t"
      "global_load_async_to_lds_b128 %0, %1, %2 offset:64\n\t"
      "global_load_async_to_lds_b128 %0, %1, %2 offset:80\n\t"
      "global_load_async_to_lds_b128 %0, %1, %2 offset:96\n\t"
      "global_load_async_to_lds_b128 %0, %1, %2 offset:112"
      :: "v"(lds), "v"(voff), "s"(g) : "memory");
}

__device__ __forceinline__ void async_wait_barrier() {
  asm volatile("s_wait_asynccnt 0x0" ::: "memory");
  __syncthreads();
}

// A-fragment (16x32 bf16) built from an LDS-resident f32 tile:
//  lanes 0-15 : row=rb+lane,    elems 0..15 = K[k0..k0+15]
//  lanes 16-31: row=rb+lane-16, elems 0..15 = K[k0+16..k0+31]
__device__ __forceinline__ bf16x16 load_a_frag_lds(const float* __restrict__ s,
                                                   int rowbase, int k0, int lane) {
  const int half = lane >> 4;
  const int l    = lane & 15;
  const float* p = s + (rowbase + l) * HDIM + (k0 + half * 16);
  bf16x16 f;
#pragma unroll
  for (int j = 0; j < 16; ++j) {
    f[j] = (bf16)p[j];
  }
  return f;
}

// Packed B fragments: tile (kt,nt), lane holds 16 contiguous bf16 (32 B).
__device__ __forceinline__ bf16x16 load_b_frag(const bf16* __restrict__ P,
                                               int kt, int nt, int lane) {
  return *reinterpret_cast<const bf16x16*>(
      P + ((((kt << 4) + nt) << 5) + lane) * 16);
}

// ---------------------------------------------------------------------------
// Weight pre-pack: W f32 [256][256] (K rows, N cols) -> bf16 WMMA B layout.
// ---------------------------------------------------------------------------
__global__ __launch_bounds__(32) void pack_weight_kernel(
    const float* __restrict__ W, bf16* __restrict__ P) {
  const int tile = blockIdx.x;       // 0..127
  const int kt = tile & 7;
  const int nt = tile >> 3;
  const int lane = threadIdx.x;
  const int half = lane >> 4;
  const int l = lane & 15;
  const int krow = kt * 32 + half * 16;
  const int col = nt * 16 + l;
  bf16x16 f;
#pragma unroll
  for (int j = 0; j < 16; ++j) f[j] = (bf16)W[(size_t)(krow + j) * HDIM + col];
  *reinterpret_cast<bf16x16*>(P + ((((kt << 4) + nt) << 5) + lane) * 16) = f;
}

// ---------------------------------------------------------------------------
// Stage 1: x_input = X@Wx + bx ; LC_raw = x_input + prev_LC@Wh + bh
// + fused per-block column stats of LC_raw.
// ---------------------------------------------------------------------------
__global__ __launch_bounds__(256) void stage1_kernel(
    const float* __restrict__ X, const float* __restrict__ PL,
    const bf16* __restrict__ PWx, const bf16* __restrict__ PWh,
    const float* __restrict__ bx, const float* __restrict__ bh,
    float* __restrict__ x_input, float* __restrict__ LC_raw,
    float* __restrict__ part) {
  __shared__ float smX[32 * HDIM];
  __shared__ float smH[32 * HDIM];
  __shared__ float sSum[2][HDIM];
  __shared__ float sSq[2][HDIM];

  const int lane = threadIdx.x & 31;
  const int w = threadIdx.x >> 5;
  const int mg = w & 1, ng = w >> 1;
  const int row0 = blockIdx.x * 32 + mg * 16;
  const int col0 = ng * 64;

  async_tile_load(X + (size_t)blockIdx.x * 32 * HDIM, smX);
  async_tile_load(PL + (size_t)blockIdx.x * 32 * HDIM, smH);
  async_wait_barrier();

  f32x8 accX[4], accH[4];
#pragma unroll
  for (int i = 0; i < 4; ++i) { accX[i] = zero8(); accH[i] = zero8(); }

#pragma unroll
  for (int kt = 0; kt < 8; ++kt) {
    bf16x16 aX = load_a_frag_lds(smX, mg * 16, kt * 32, lane);
    bf16x16 aH = load_a_frag_lds(smH, mg * 16, kt * 32, lane);
#pragma unroll
    for (int i = 0; i < 4; ++i) {
      const int nt = ng * 4 + i;
      accX[i] = wmma_bf16(aX, load_b_frag(PWx, kt, nt, lane), accX[i]);
      accH[i] = wmma_bf16(aH, load_b_frag(PWh, kt, nt, lane), accH[i]);
    }
  }

  const int half = lane >> 4, l = lane & 15;
#pragma unroll
  for (int i = 0; i < 4; ++i) {
    const int col = col0 + i * 16 + l;
    const float bxv = bx[col], bhv = bh[col];
    const size_t base = (size_t)(row0 + half * 8) * HDIM + col;
    float sr = 0.0f, sq = 0.0f;
#pragma unroll
    for (int r = 0; r < 8; ++r) {
      const float xv = accX[i][r] + bxv;
      const float lc = xv + accH[i][r] + bhv;
      x_input[base + (size_t)r * HDIM] = xv;
      LC_raw[base + (size_t)r * HDIM] = lc;
      sr += lc; sq += lc * lc;
    }
    sr += __shfl_down(sr, 16);
    sq += __shfl_down(sq, 16);
    if (lane < 16) { sSum[mg][col] = sr; sSq[mg][col] = sq; }
  }
  __syncthreads();
  const int t = threadIdx.x;
  part[(size_t)blockIdx.x * 512 + t]       = sSum[0][t] + sSum[1][t];
  part[(size_t)blockIdx.x * 512 + 256 + t] = sSq[0][t] + sSq[1][t];
}

// ---------------------------------------------------------------------------
// Generic single GEMM. MODE 0: out = A@W + bias, + fused column stats.
//                      MODE 1: out = sigmoid(A@W + bias) * srow[row] (Pupil).
// ---------------------------------------------------------------------------
template <int MODE>
__global__ __launch_bounds__(256) void gemm256_kernel(
    const float* __restrict__ A, const bf16* __restrict__ PW,
    const float* __restrict__ bias, const float* __restrict__ srow,
    float* __restrict__ out, float* __restrict__ part) {
  __shared__ float smA[32 * HDIM];
  __shared__ float sSum[2][HDIM];
  __shared__ float sSq[2][HDIM];

  const int lane = threadIdx.x & 31;
  const int w = threadIdx.x >> 5;
  const int mg = w & 1, ng = w >> 1;
  const int row0 = blockIdx.x * 32 + mg * 16;
  const int col0 = ng * 64;

  async_tile_load(A + (size_t)blockIdx.x * 32 * HDIM, smA);
  async_wait_barrier();

  f32x8 acc[4];
#pragma unroll
  for (int i = 0; i < 4; ++i) acc[i] = zero8();

#pragma unroll
  for (int kt = 0; kt < 8; ++kt) {
    bf16x16 a = load_a_frag_lds(smA, mg * 16, kt * 32, lane);
#pragma unroll
    for (int i = 0; i < 4; ++i)
      acc[i] = wmma_bf16(a, load_b_frag(PW, kt, ng * 4 + i, lane), acc[i]);
  }

  const int half = lane >> 4, l = lane & 15;
#pragma unroll
  for (int i = 0; i < 4; ++i) {
    const int col = col0 + i * 16 + l;
    const float bv = bias[col];
    float sr = 0.0f, sq = 0.0f;
#pragma unroll
    for (int r = 0; r < 8; ++r) {
      const int row = row0 + half * 8 + r;
      float v = acc[i][r] + bv;
      if (MODE == 1) v = sigmoidf_(v) * srow[row];
      out[(size_t)row * HDIM + col] = v;
      sr += v; sq += v * v;
    }
    if (MODE == 0) {
      sr += __shfl_down(sr, 16);
      sq += __shfl_down(sq, 16);
      if (lane < 16) { sSum[mg][col] = sr; sSq[mg][col] = sq; }
    }
  }
  if (MODE == 0) {
    __syncthreads();
    const int t = threadIdx.x;
    part[(size_t)blockIdx.x * 512 + t]       = sSum[0][t] + sSum[1][t];
    part[(size_t)blockIdx.x * 512 + 256 + t] = sSq[0][t] + sSq[1][t];
  }
}

// ---------------------------------------------------------------------------
// Cell kernel: 4 fused GEMMs over a shared row tile (wave tile 16x32).
// new_cell = sig(PC@Wf+bf)*cell + sig(XI@Wi+bi)*0.1*(NE@WLC+bLC) + XI@WC+bC
// + fused per-block column stats of new_cell (block covers 128 cols).
// ---------------------------------------------------------------------------
__global__ __launch_bounds__(256) void cell_kernel(
    const float* __restrict__ NEt, const float* __restrict__ PC,
    const float* __restrict__ XI, const float* __restrict__ CS,
    const bf16* __restrict__ PWLC, const bf16* __restrict__ PWf,
    const bf16* __restrict__ PWi, const bf16* __restrict__ PWC,
    const float* __restrict__ bLC, const float* __restrict__ bf_,
    const float* __restrict__ bi_, const float* __restrict__ bC_,
    float* __restrict__ new_cell, float* __restrict__ part) {
  __shared__ float smN[32 * HDIM];
  __shared__ float smP[32 * HDIM];
  __shared__ float smX[32 * HDIM];
  __shared__ float sSum[2][128];
  __shared__ float sSq[2][128];

  const int lane = threadIdx.x & 31;
  const int w = threadIdx.x >> 5;
  const int mg = w & 1, ng = w >> 1;
  const int row0 = blockIdx.x * 32 + mg * 16;
  const int colbase = blockIdx.y * 128 + ng * 32;
  const int nt0 = colbase >> 4;

  async_tile_load(NEt + (size_t)blockIdx.x * 32 * HDIM, smN);
  async_tile_load(PC + (size_t)blockIdx.x * 32 * HDIM, smP);
  async_tile_load(XI + (size_t)blockIdx.x * 32 * HDIM, smX);
  async_wait_barrier();

  f32x8 accN[2], accF[2], accI[2], accC[2];
#pragma unroll
  for (int i = 0; i < 2; ++i) {
    accN[i] = zero8(); accF[i] = zero8(); accI[i] = zero8(); accC[i] = zero8();
  }

#pragma unroll
  for (int kt = 0; kt < 8; ++kt) {
    bf16x16 aN = load_a_frag_lds(smN, mg * 16, kt * 32, lane);
    bf16x16 aP = load_a_frag_lds(smP, mg * 16, kt * 32, lane);
    bf16x16 aX = load_a_frag_lds(smX, mg * 16, kt * 32, lane);
#pragma unroll
    for (int i = 0; i < 2; ++i) {
      const int nt = nt0 + i;
      accN[i] = wmma_bf16(aN, load_b_frag(PWLC, kt, nt, lane), accN[i]);
      accF[i] = wmma_bf16(aP, load_b_frag(PWf, kt, nt, lane), accF[i]);
      accI[i] = wmma_bf16(aX, load_b_frag(PWi, kt, nt, lane), accI[i]);
      accC[i] = wmma_bf16(aX, load_b_frag(PWC, kt, nt, lane), accC[i]);
    }
  }

  const int half = lane >> 4, l = lane & 15;
#pragma unroll
  for (int i = 0; i < 2; ++i) {
    const int col = colbase + i * 16 + l;
    const int lc = ng * 32 + i * 16 + l;   // col local to this block (0..127)
    const float blcv = bLC[col], bfv = bf_[col], biv = bi_[col], bcv = bC_[col];
    float sr = 0.0f, sq = 0.0f;
#pragma unroll
    for (int r = 0; r < 8; ++r) {
      const int row = row0 + half * 8 + r;
      const size_t idx = (size_t)row * HDIM + col;
      const float fg = sigmoidf_(accF[i][r] + bfv);
      const float ig = sigmoidf_(accI[i][r] + biv);
      const float nc = fg * CS[idx] + ig * 0.1f * (accN[i][r] + blcv)
                       + accC[i][r] + bcv;
      new_cell[idx] = nc;
      sr += nc; sq += nc * nc;
    }
    sr += __shfl_down(sr, 16);
    sq += __shfl_down(sq, 16);
    if (lane < 16) { sSum[mg][lc] = sr; sSq[mg][lc] = sq; }
  }
  __syncthreads();
  const int t = threadIdx.x;
  if (t < 128) {
    const size_t o = (size_t)blockIdx.x * 512 + blockIdx.y * 128 + t;
    part[o]       = sSum[0][t] + sSum[1][t];
    part[o + 256] = sSq[0][t] + sSq[1][t];
  }
}

// ---------------------------------------------------------------------------
// Stats finalize (deterministic serial combine) + BN scale/shift.
// ---------------------------------------------------------------------------
__global__ __launch_bounds__(256) void colstats_final_kernel(
    const float* __restrict__ part, const float* __restrict__ gamma,
    const float* __restrict__ beta, float* __restrict__ scale,
    float* __restrict__ shift, int nblk) {
  const int col = threadIdx.x;
  float s = 0.0f, q = 0.0f;
  for (int b = 0; b < nblk; ++b) {
    s += part[(size_t)b * 512 + col];
    q += part[(size_t)b * 512 + 256 + col];
  }
  const float inv = 1.0f / 65536.0f;
  const float mean = s * inv;
  const float var = q * inv - mean * mean;
  const float sc = gamma[col] * rsqrtf(var + 1e-5f);
  scale[col] = sc;
  shift[col] = beta[col] - mean * sc;
}

__global__ __launch_bounds__(256) void bn_leaky_kernel(
    const float* __restrict__ x, const float* __restrict__ scale,
    const float* __restrict__ shift, float* __restrict__ y) {
  const size_t i = ((size_t)blockIdx.x * 256 + threadIdx.x) * 4;
  const int col = (int)(i & 255);
  float4 v = *reinterpret_cast<const float4*>(x + i);
  float o0 = v.x * scale[col + 0] + shift[col + 0];
  float o1 = v.y * scale[col + 1] + shift[col + 1];
  float o2 = v.z * scale[col + 2] + shift[col + 2];
  float o3 = v.w * scale[col + 3] + shift[col + 3];
  o0 = o0 > 0.0f ? o0 : 0.1f * o0;
  o1 = o1 > 0.0f ? o1 : 0.1f * o1;
  o2 = o2 > 0.0f ? o2 : 0.1f * o2;
  o3 = o3 > 0.0f ? o3 : 0.1f * o3;
  *reinterpret_cast<float4*>(y + i) = make_float4(o0, o1, o2, o3);
}

// s[b] = dot(LC_t[b]+NE_t[b]+C_t[b], WP) + bP  (one wave per row)
__global__ __launch_bounds__(256) void rowdot_kernel(
    const float* __restrict__ LC, const float* __restrict__ NE,
    const float* __restrict__ C, const float* __restrict__ WP,
    const float* __restrict__ bP, float* __restrict__ srow) {
  const int lane = threadIdx.x & 31;
  const int wv = threadIdx.x >> 5;
  const int row = blockIdx.x * 8 + wv;
  const size_t base = (size_t)row * HDIM;
  float sum = 0.0f;
#pragma unroll
  for (int j = lane; j < HDIM; j += 32)
    sum += (LC[base + j] + NE[base + j] + C[base + j]) * WP[j];
#pragma unroll
  for (int off = 16; off > 0; off >>= 1) sum += __shfl_down(sum, off, 32);
  if (lane == 0) srow[row] = sum + bP[0];
}

// ---------------------------------------------------------------------------
extern "C" void kernel_launch(void* const* d_in, const int* in_sizes, int n_in,
                              void* d_out, int out_size, void* d_ws,
                              size_t ws_size, hipStream_t stream) {
  (void)in_sizes; (void)n_in; (void)out_size; (void)ws_size;

  const float* X    = (const float*)d_in[0];
  const float* pLC  = (const float*)d_in[1];
  const float* pCx  = (const float*)d_in[2];
  const float* cell = (const float*)d_in[3];
  const float* Wx  = (const float*)d_in[4];  const float* bx  = (const float*)d_in[5];
  const float* Wh  = (const float*)d_in[6];  const float* bh  = (const float*)d_in[7];
  const float* WLC = (const float*)d_in[8];  const float* bLC = (const float*)d_in[9];
  const float* WC  = (const float*)d_in[10]; const float* bC  = (const float*)d_in[11];
  const float* WP  = (const float*)d_in[12]; const float* bP  = (const float*)d_in[13];
  const float* Wf  = (const float*)d_in[14]; const float* bf_ = (const float*)d_in[15];
  const float* Wi  = (const float*)d_in[16]; const float* bi_ = (const float*)d_in[17];
  const float* Wo  = (const float*)d_in[18]; const float* bo  = (const float*)d_in[19];
  const float* g1 = (const float*)d_in[20]; const float* be1 = (const float*)d_in[21];
  const float* g2 = (const float*)d_in[22]; const float* be2 = (const float*)d_in[23];
  const float* g3 = (const float*)d_in[24]; const float* be3 = (const float*)d_in[25];

  float* outLC   = (float*)d_out;            // LC_t (LC_raw first)
  float* outNE   = outLC + BH;               // NE_t (NE_raw first)
  float* outC    = outLC + 2 * BH;           // C_t
  float* outP    = outLC + 3 * BH;           // Pupil_t
  float* outCell = outLC + 4 * BH;           // new_cell

  // Workspace layout
  bf16* PWx  = (bf16*)d_ws;            // 7 x 65536 bf16 packed weights
  bf16* PWh  = PWx + 65536;
  bf16* PWLC = PWx + 2 * 65536;
  bf16* PWC  = PWx + 3 * 65536;
  bf16* PWf  = PWx + 4 * 65536;
  bf16* PWi  = PWx + 5 * 65536;
  bf16* PWo  = PWx + 6 * 65536;
  float* ss      = (float*)(PWx + 7 * 65536); // 3 x (scale256, shift256)
  float* srow    = ss + 3 * 512;              // 65536 floats
  float* part    = srow + 65536;              // 2048*512 floats
  float* x_input = part + 2048 * 512;         // BH floats

  // --- 1. pack weights to bf16 WMMA B-fragment layout ---
  pack_weight_kernel<<<128, 32, 0, stream>>>(Wx, PWx);
  pack_weight_kernel<<<128, 32, 0, stream>>>(Wh, PWh);
  pack_weight_kernel<<<128, 32, 0, stream>>>(WLC, PWLC);
  pack_weight_kernel<<<128, 32, 0, stream>>>(WC, PWC);
  pack_weight_kernel<<<128, 32, 0, stream>>>(Wf, PWf);
  pack_weight_kernel<<<128, 32, 0, stream>>>(Wi, PWi);
  pack_weight_kernel<<<128, 32, 0, stream>>>(Wo, PWo);

  const int gGemm = NROWS / 32;   // 2048
  const int gElem = (int)(BH / 4 / 256);

  // --- 2. x_input and LC_raw (+ fused BN1 stats) ---
  stage1_kernel<<<gGemm, 256, 0, stream>>>(X, pLC, PWx, PWh, bx, bh,
                                           x_input, outLC, part);
  colstats_final_kernel<<<1, 256, 0, stream>>>(part, g1, be1, ss, ss + 256,
                                               gGemm);
  bn_leaky_kernel<<<gElem, 256, 0, stream>>>(outLC, ss, ss + 256, outLC);

  // --- 3. NE_raw = LC_t@WLC + bLC (+ fused BN2 stats) ; bn+leaky -> NE_t ---
  gemm256_kernel<0><<<gGemm, 256, 0, stream>>>(outLC, PWLC, bLC, nullptr,
                                               outNE, part);
  colstats_final_kernel<<<1, 256, 0, stream>>>(part, g2, be2, ss + 512,
                                               ss + 768, gGemm);
  bn_leaky_kernel<<<gElem, 256, 0, stream>>>(outNE, ss + 512, ss + 768, outNE);

  // --- 4. fused gates + new_cell (+ fused BN3 stats) ---
  cell_kernel<<<dim3(gGemm, 2), 256, 0, stream>>>(
      outNE, pCx, x_input, cell, PWLC, PWf, PWi, PWC, bLC, bf_, bi_, bC,
      outCell, part);
  colstats_final_kernel<<<1, 256, 0, stream>>>(part, g3, be3, ss + 1024,
                                               ss + 1280, gGemm);
  bn_leaky_kernel<<<gElem, 256, 0, stream>>>(outCell, ss + 1024, ss + 1280,
                                             outC);

  // --- 5. Pupil = sigmoid(C_t@Wo+bo) * ((LC+NE+C)@WP + bP) ---
  rowdot_kernel<<<NROWS / 8, 256, 0, stream>>>(outLC, outNE, outC, WP, bP, srow);
  gemm256_kernel<1><<<gGemm, 256, 0, stream>>>(outC, PWo, bo, srow, outP,
                                               nullptr);
}
